// Convolution_77111842832763
// MI455X (gfx1250) — matrix operand order repaired and checked by reading, not compiled, not run on hardware
//
#include <hip/hip_runtime.h>
#include <hip/hip_bf16.h>
#include <stdint.h>

typedef __attribute__((ext_vector_type(16))) __bf16 v16bf;
typedef __attribute__((ext_vector_type(8)))  float  v8f;

union Frag { v16bf v; int4 q[2]; };

__device__ __forceinline__ unsigned short f2bf(float f) {
    union { float f; uint32_t u; } c; c.f = f;
    uint32_t u = c.u;
    uint32_t r = (u + 0x7FFFu + ((u >> 16) & 1u)) >> 16;   // RNE
    return (unsigned short)r;
}

// ---------------------------------------------------------------------------
// prep_x: x (2,40,40,40,64) f32  ->  xp (2,44,44,52,64) bf16, zero padded.
// ---------------------------------------------------------------------------
__global__ void prep_x(const float* __restrict__ x, uint32_t* __restrict__ xp, int total)
{
    int id = blockIdx.x * blockDim.x + threadIdx.x;
    if (id >= total) return;
    int pc = id & 31;
    int pt = id >> 5;
    int pz = pt % 52; pt /= 52;
    int py = pt % 44; pt /= 44;
    int px = pt % 44; int n = pt / 44;
    uint32_t val = 0;
    if (px >= 2 && px < 42 && py >= 2 && py < 42 && pz >= 2 && pz < 42) {
        size_t s = ((((size_t)n*40 + (px-2))*40 + (py-2))*40 + (pz-2))*64 + (size_t)pc*2;
        float f0 = x[s], f1 = x[s+1];
        val = (uint32_t)f2bf(f0) | ((uint32_t)f2bf(f1) << 16);
    }
    xp[id] = val;
}

// ---------------------------------------------------------------------------
// prep_w: build 125 tap matrices (tp_right of emb@weight/125), fold the
// self-connection into center tap l==62, emit bf16 in per-lane WMMA
// B-fragment order: wpre[l][kc][nt][lane][j],
//   cin = kc*32 + (lane>>4)*16 + j,  cout = nt*16 + (lane&15)
// ---------------------------------------------------------------------------
__global__ void prep_w(const float* __restrict__ lw, const float* __restrict__ weight,
                       unsigned short* __restrict__ wpre)
{
    __shared__ float wl[1024];
    int l   = blockIdx.x;
    int tid = threadIdx.x;
    int iz = l % 5, iy = (l/5) % 5, ix = l / 25;
    float vx = ix - 2.0f, vy = iy - 2.0f, vz = iz - 2.0f;
    float r = sqrtf(vx*vx + vy*vy + vz*vz);

    float emb[8];
    #pragma unroll
    for (int b = 0; b < 8; ++b) {
        float diff = r * 3.6f - (float)(b + 1);            // step = 2.5/9
        float den  = fmaxf(1.0f - diff*diff, 1e-9f);
        emb[b] = (fabsf(diff) < 1.0f) ? 1.14136f * __expf(2.0f - 1.0f/den) : 0.0f;
    }
    float y1[3] = {0.f, 0.f, 0.f};
    if (r > 0.f) { float s = 1.7320508075688772f / r; y1[0]=vx*s; y1[1]=vy*s; y1[2]=vz*s; }

    #pragma unroll
    for (int j4 = 0; j4 < 4; ++j4) {
        int j = tid*4 + j4;
        float acc = 0.f;
        #pragma unroll
        for (int b = 0; b < 8; ++b) acc += emb[b] * weight[b*1024 + j];
        wl[j] = acc * (1.0f/125.0f);
    }
    __syncthreads();

    const float a  = 0.17677669529663687f;                 // 1/sqrt(2*MUL)
    const float a3 = a * 0.5773502691896258f;              // a/sqrt(3)

    for (int i = 0; i < 16; ++i) {
        int t    = tid + 256*i;                            // 0..4095
        int j    = t & 15;
        int lane = (t >> 4) & 31;
        int nt   = (t >> 9) & 3;
        int kc   = (t >> 11) & 1;
        int cin  = kc*32 + (lane >> 4)*16 + j;
        int cout = nt*16 + (lane & 15);
        float val;
        if (cin < 16) {
            int u = cin;
            if (cout < 16) {
                val = a * wl[0*256 + u*16 + cout];
                if (l == 62) val += 0.25f * lw[0*256 + u*16 + cout];
            } else {
                int cv = cout - 16; int w = cv/3; int m = cv%3;
                val = a * wl[256 + u*16 + w] * y1[m];
            }
        } else {
            int ci = cin - 16; int u = ci/3; int m = ci%3;
            if (cout < 16) {
                val = a3 * wl[768 + u*16 + cout] * y1[m];
            } else {
                int cv = cout - 16; int w = cv/3; int mm = cv%3;
                val = (m == mm) ? a * wl[512 + u*16 + w] : 0.0f;
                if (l == 62 && m == mm) val += 0.25f * lw[256 + u*16 + w];
            }
        }
        wpre[(size_t)l*4096 + t] = f2bf(val);
    }
}

// ---------------------------------------------------------------------------
// conv_wmma: 600 workgroups (n=2 x 10 xb x 10 yb x 3 zb), 8 waves.
// Wave = two (X,Y) rows x 16 z x 64 couts  (M=32 total, B reused 2x).
// Weights: 40KB/stage, double-buffered async-to-LDS, overlapped with compute.
// ---------------------------------------------------------------------------
__global__ void __launch_bounds__(256) conv_wmma(const uint8_t* __restrict__ xp,
                                                 const uint8_t* __restrict__ wpre,
                                                 float* __restrict__ out)
{
    __shared__ __align__(16) uint8_t wbuf[2][40960];       // 2 x (5 taps x 8KB)
    int tid  = threadIdx.x;
    int lane = tid & 31;
    int wave = tid >> 5;
    int m = lane & 15, h = lane >> 4;

    int bid = blockIdx.x;
    int zb = bid % 3;  bid /= 3;
    int yb = bid % 10; bid /= 10;
    int xb = bid % 10; int n = bid / 10;
    int rx  = wave >> 1;
    int ryb = (wave & 1) * 2;
    int X  = xb*4 + rx;
    int Y0 = yb*4 + ryb;                                   // rows Y0, Y0+1
    int z0 = zb*16;

    uint32_t lb0 = (uint32_t)(uintptr_t)(void*)&wbuf[0][0];

    v8f acc[2][4] = {};

    // prologue: stage 0 -> buffer 0
    #pragma unroll
    for (int i = 0; i < 10; ++i) {
        uint32_t chunk = (uint32_t)tid + 256u*i;
        uint32_t voff  = chunk*16u;
        uint32_t loff  = lb0 + chunk*16u;
        asm volatile("global_load_async_to_lds_b128 %0, %1, %2"
                     :: "v"(loff), "v"(voff), "s"(wpre) : "memory");
    }

    for (int dxy = 0; dxy < 25; ++dxy) {
        int dx = dxy / 5, dy = dxy % 5;
        int p  = dxy & 1;

        if (dxy + 1 < 25) {                                // stage dxy+1 -> buffer p^1
            uint32_t lbn = lb0 + (uint32_t)(p ^ 1)*40960u;
            #pragma unroll
            for (int i = 0; i < 10; ++i) {
                uint32_t chunk = (uint32_t)tid + 256u*i;
                uint32_t voff  = (uint32_t)(dxy + 1)*40960u + chunk*16u;
                uint32_t loff  = lbn + chunk*16u;
                asm volatile("global_load_async_to_lds_b128 %0, %1, %2"
                             :: "v"(loff), "v"(voff), "s"(wpre) : "memory");
            }
            // allow the 10 just-issued chunks to stay in flight; older ones done
            asm volatile("s_wait_asynccnt 0xa" ::: "memory");
        } else {
            asm volatile("s_wait_asynccnt 0x0" ::: "memory");
        }
        __syncthreads();                                   // stage dxy visible

        const uint8_t* row0 =
            xp + ((((size_t)n*44 + (X + dx))*44 + (Y0 + dy))*52) * 128;
        const uint8_t* row1 = row0 + 52*128;               // Y0+1
        const uint8_t* wb   = &wbuf[p][0] + (size_t)lane*32;

        // prefetch next stage's input rows a full stage early
        if (dxy + 1 < 25 && lane < 20) {
            int dxn = (dxy + 1)/5, dyn = (dxy + 1)%5;
            const uint8_t* nr =
                xp + ((((size_t)n*44 + (X + dxn))*44 + (Y0 + dyn))*52) * 128
                   + (size_t)(z0 + lane)*128;
            __builtin_prefetch(nr, 0, 1);
            __builtin_prefetch(nr + 52*128, 0, 1);
        }

        #pragma unroll
        for (int dz = 0; dz < 5; ++dz) {
            const uint8_t* p0 = row0 + (size_t)(z0 + m + dz)*128 + h*32;
            const uint8_t* p1 = row1 + (size_t)(z0 + m + dz)*128 + h*32;
            #pragma unroll
            for (int kc = 0; kc < 2; ++kc) {
                Frag Ar0, Ar1;                             // contiguous k = 16h+s per k-chunk
                Ar0.q[0] = *(const int4*)(p0 + kc*64);
                Ar0.q[1] = *(const int4*)(p0 + kc*64 + 16);
                Ar1.q[0] = *(const int4*)(p1 + kc*64);
                Ar1.q[1] = *(const int4*)(p1 + kc*64 + 16);
                #pragma unroll
                for (int nt = 0; nt < 4; ++nt) {
                    Frag B = *(const Frag*)(wb + (size_t)(dz*8 + kc*4 + nt)*1024);
                    acc[0][nt] = __builtin_amdgcn_wmma_f32_16x16x32_bf16(
                                     false, Ar0.v, false, B.v, (short)0, acc[0][nt], false, false);
                    acc[1][nt] = __builtin_amdgcn_wmma_f32_16x16x32_bf16(
                                     false, Ar1.v, false, B.v, (short)0, acc[1][nt], false, false);
                }
            }
        }
        __syncthreads();                                   // buf p free for reuse
    }

    // C/D layout: VGPR r, lane -> M = r + 8*h (z), N = lane&15 (cout in tile)
    #pragma unroll
    for (int s = 0; s < 2; ++s) {
        float* outp = out + ((((size_t)n*40 + X)*40 + (Y0 + s))*40) * 64;
        #pragma unroll
        for (int nt = 0; nt < 4; ++nt) {
            #pragma unroll
            for (int r = 0; r < 8; ++r) {
                int z = z0 + r + 8*h;
                if (z < 40) outp[(size_t)z*64 + nt*16 + m] = acc[s][nt][r];
            }
        }
    }
}

// ---------------------------------------------------------------------------
extern "C" void kernel_launch(void* const* d_in, const int* in_sizes, int n_in,
                              void* d_out, int out_size, void* d_ws, size_t ws_size,
                              hipStream_t stream)
{
    const float* x      = (const float*)d_in[0];
    const float* lw     = (const float*)d_in[1];   // linear_weight (512)
    const float* weight = (const float*)d_in[2];   // (8,1024)

    uint8_t* ws   = (uint8_t*)d_ws;
    uint8_t* xp   = ws;                                        // 25,776,128 B
    unsigned short* wpre = (unsigned short*)(ws + 25776128);   // 1,024,000 B

    int totalPairs = 2*44*44*52*32;
    prep_x<<<(totalPairs + 255)/256, 256, 0, stream>>>(x, (uint32_t*)xp, totalPairs);
    prep_w<<<125, 256, 0, stream>>>(lw, weight, wpre);
    conv_wmma<<<600, 256, 0, stream>>>(xp, (const uint8_t*)wpre, (float*)d_out);
}
